// OutlookAttention_83124797047087
// MI455X (gfx1250) — compile-verified, matrix operand order checked by reading, not run on hardware
//
#include <hip/hip_runtime.h>
#include <hip/hip_bf16.h>

// ---------------------------------------------------------------------------
// OutlookAttention (VOLO) for MI455X / gfx1250, wave32 + WMMA bf16 pipeline.
//   Three GEMMs (M=50176, K=384, N=54/384/384) as v_wmma_f32_16x16x32_bf16.
//   v3: B-tile staged in LDS (49KB of 320KB WGP LDS, +8 bf16 row pad ->
//   conflict-free ds_load_b128) and the K-loop is explicitly double-buffered:
//   next K-step's 2 A-frags (global) + 4 B-frags (LDS) are issued before the
//   current step's 8 WMMAs, so s_wait_{loadcnt,dscnt} land on data that had a
//   whole WMMA batch to arrive instead of stalling on just-issued loads.
// ---------------------------------------------------------------------------

typedef __attribute__((ext_vector_type(16))) __bf16          v16bf;
typedef __attribute__((ext_vector_type(8)))  float           v8f;
typedef __attribute__((ext_vector_type(8)))  unsigned short  us8;   // 16B
typedef __attribute__((ext_vector_type(4)))  unsigned short  us4;   // 8B

static constexpr int CDIM  = 384;
static constexpr int NHEAD = 6;
static constexpr int BATCH = 16;
static constexpr int HH    = 56;
static constexpr int WW    = 56;
static constexpr int PIX   = BATCH * HH * WW;          // 50176
static constexpr long XELEMS = (long)PIX * CDIM;       // 19,267,584
static constexpr int LDSB_STRIDE = CDIM + 8;           // 392: bank stride 4 -> conflict-free

__device__ __forceinline__ unsigned short f2bf(float f) {
  unsigned u = __builtin_bit_cast(unsigned, f);
  u += 0x7FFFu + ((u >> 16) & 1u);                     // round-to-nearest-even
  return (unsigned short)(u >> 16);
}
__device__ __forceinline__ float bf2f(unsigned short h) {
  return __builtin_bit_cast(float, (unsigned)h << 16);
}

// -------------------------- pack kernels -----------------------------------
__global__ __launch_bounds__(256) void pack_x(const float* __restrict__ x,
                                              unsigned short* __restrict__ xb,
                                              long n4) {
  long t = (long)blockIdx.x * blockDim.x + threadIdx.x;
  if (t >= n4) return;
  const float4 f = ((const float4*)x)[t];
  us4 o; o[0] = f2bf(f.x); o[1] = f2bf(f.y); o[2] = f2bf(f.z); o[3] = f2bf(f.w);
  *(us4*)(xb + t * 4) = o;
}

// W[in=384, out] (row-major) -> Wt[out_pad, 384] bf16, zero-padded rows.
__global__ __launch_bounds__(256) void pack_wt(const float* __restrict__ W,
                                               unsigned short* __restrict__ Wt,
                                               int opad, int oreal, int ldo) {
  int t = blockIdx.x * blockDim.x + threadIdx.x;
  if (t >= opad * CDIM) return;
  int o = t / CDIM, i = t - o * CDIM;
  Wt[(size_t)o * CDIM + i] = (o < oreal) ? f2bf(W[(size_t)i * ldo + o]) : 0;
}

// -------------------------- WMMA GEMM --------------------------------------
// 16-bit A/B fragment layout (ISA 7.12.2): lane%16 = row (A) / col (B);
// lanes 0-15 hold K {0..7,16..23}, lanes 16-31 hold K {8..15,24..31}.
__device__ __forceinline__ v16bf load_fragA(const unsigned short* __restrict__ base,
                                            int row, int k0, int lane) {
  const int kb = k0 + ((lane & 16) ? 8 : 0);
  const unsigned short* p = base + (size_t)row * CDIM + kb;
  union { v16bf v; us8 h[2]; } f;
  f.h[0] = *(const us8*)(p);        // K = kb .. kb+7
  f.h[1] = *(const us8*)(p + 16);   // K = kb+16 .. kb+23
  return f.v;
}
__device__ __forceinline__ v16bf load_fragB_lds(const unsigned short* lds,
                                                int rowInTile, int k0, int lane) {
  const int kb = k0 + ((lane & 16) ? 8 : 0);
  const unsigned short* p = lds + rowInTile * LDSB_STRIDE + kb;
  union { v16bf v; us8 h[2]; } f;
  f.h[0] = *(const us8*)(p);        // ds_load_b128
  f.h[1] = *(const us8*)(p + 16);
  return f.v;
}

// C = A[50176 x 384](bf16) @ Bt^T (Bt stored [Npad x 384] bf16) + bias.
// Block = 256 thr = 8 waves; wave tile = 32(M) x 64(N); grid = (196, Npad/64).
template <bool OUT_BF16>
__global__ __launch_bounds__(256) void gemm384_wmma(
    const unsigned short* __restrict__ A, const unsigned short* __restrict__ Bt,
    const float* __restrict__ bias, void* __restrict__ outp, int nreal, int ldo) {
  __shared__ unsigned short ldsB[64 * LDSB_STRIDE];    // 50,176 B

  const int tid  = threadIdx.x;
  const int lane = tid & 31;
  const int wave = tid >> 5;
  const int m0 = (blockIdx.x * 8 + wave) * 32;
  const int n0 = blockIdx.y * 64;

  // ---- stage Bt[n0..n0+63][0..383] -> LDS (12 x b128 per thread) ----
#pragma unroll
  for (int it = 0; it < 12; ++it) {
    const int chunk = it * 256 + tid;                  // 8 bf16 per chunk
    const int row = (chunk * 8) / CDIM;
    const int col = (chunk * 8) % CDIM;
    *(us8*)(ldsB + row * LDSB_STRIDE + col) =
        *(const us8*)(Bt + (size_t)(n0 + row) * CDIM + col);
  }
  __syncthreads();

  v8f c[2][4];
#pragma unroll
  for (int a = 0; a < 2; ++a)
#pragma unroll
    for (int j = 0; j < 4; ++j)
#pragma unroll
      for (int e = 0; e < 8; ++e) c[a][j][e] = 0.0f;

  const int rA = m0 + (lane & 15);
  const int rB = lane & 15;

  // ---- software-pipelined K loop (double-buffered fragments) ----
  v16bf a0 = load_fragA(A, rA, 0, lane);
  v16bf a1 = load_fragA(A, rA + 16, 0, lane);
  v16bf b[4];
#pragma unroll
  for (int j = 0; j < 4; ++j) b[j] = load_fragB_lds(ldsB, rB + 16 * j, 0, lane);

#pragma unroll
  for (int k0 = 0; k0 < CDIM; k0 += 32) {
    v16bf na0 = a0, na1 = a1, nb[4] = {b[0], b[1], b[2], b[3]};
    if (k0 + 32 < CDIM) {                              // issue next-step loads first
      na0 = load_fragA(A, rA, k0 + 32, lane);
      na1 = load_fragA(A, rA + 16, k0 + 32, lane);
#pragma unroll
      for (int j = 0; j < 4; ++j) nb[j] = load_fragB_lds(ldsB, rB + 16 * j, k0 + 32, lane);
    }
#pragma unroll
    for (int j = 0; j < 4; ++j) {
      c[0][j] = __builtin_amdgcn_wmma_f32_16x16x32_bf16(
          false, a0, false, b[j], (short)0, c[0][j], false, false);
      c[1][j] = __builtin_amdgcn_wmma_f32_16x16x32_bf16(
          false, a1, false, b[j], (short)0, c[1][j], false, false);
    }
    a0 = na0; a1 = na1;
#pragma unroll
    for (int j = 0; j < 4; ++j) b[j] = nb[j];
  }

  // D layout: VGPR r -> M = r + 8*(lane/16), N = lane%16 (ISA 7.12.2)
  float* outF = (float*)outp;
  unsigned short* outB = (unsigned short*)outp;
#pragma unroll
  for (int ai = 0; ai < 2; ++ai) {
#pragma unroll
    for (int j = 0; j < 4; ++j) {
      const int ncol = n0 + j * 16 + (lane & 15);
      if (ncol >= nreal) continue;
      const float bv = bias[ncol];
      const int mbase = m0 + ai * 16 + ((lane >> 4) << 3);
#pragma unroll
      for (int r = 0; r < 8; ++r) {
        const float val = c[ai][j][r] + bv;
        const size_t idx = (size_t)(mbase + r) * ldo + ncol;
        if (OUT_BF16) outB[idx] = f2bf(val);
        else          outF[idx] = val;
      }
    }
  }
}

// -------------------------- softmax over 3x3 window -------------------------
// attn layout [pixel][head*9+w]; one thread per (pixel,head).
__global__ __launch_bounds__(256) void softmax9(float* __restrict__ a, int nph) {
  int t = blockIdx.x * blockDim.x + threadIdx.x;
  if (t >= nph) return;
  float* p = a + (size_t)t * 9;
  float m = p[0], e[9];
#pragma unroll
  for (int w = 1; w < 9; ++w) m = fmaxf(m, p[w]);
  float s = 0.f;
#pragma unroll
  for (int w = 0; w < 9; ++w) { e[w] = __expf(p[w] - m); s += e[w]; }
  const float inv = 1.0f / s;
#pragma unroll
  for (int w = 0; w < 9; ++w) p[w] = e[w] * inv;
}

// ------------------- softmax-weighted 3x3 gather (y) ------------------------
// 8 channels / thread, b128 loads; v (38MB bf16) is L2-resident so the 9x
// re-read stays on-chip.  8 | head_dim(64) so one head per thread.
__global__ __launch_bounds__(256) void aggregate(const float* __restrict__ attn,
                                                 const unsigned short* __restrict__ v,
                                                 unsigned short* __restrict__ y) {
  long t = (long)blockIdx.x * blockDim.x + threadIdx.x;   // over XELEMS/8
  if (t >= XELEMS / 8) return;
  const int c8 = (int)(t % (CDIM / 8)) * 8;
  const long p = t / (CDIM / 8);
  const int b = (int)(p / (HH * WW));
  const int rem = (int)(p % (HH * WW));
  const int i = rem / WW, j = rem % WW;
  const int h = c8 / (CDIM / NHEAD);                      // head dim = 64
  const float* aw = attn + ((size_t)p * NHEAD + h) * 9;

  float acc[8];
#pragma unroll
  for (int e = 0; e < 8; ++e) acc[e] = 0.f;
#pragma unroll
  for (int w = 0; w < 9; ++w) {                           // (ki,kj) row-major
    const int ii = i + w / 3 - 1, jj = j + w % 3 - 1;
    if (ii >= 0 && ii < HH && jj >= 0 && jj < WW) {
      const size_t q = ((size_t)b * HH * WW + (size_t)ii * WW + jj) * CDIM + c8;
      const us8 vv = *(const us8*)(v + q);
      const float a = aw[w];
#pragma unroll
      for (int e = 0; e < 8; ++e) acc[e] = fmaf(a, bf2f(vv[e]), acc[e]);
    }
  }
  us8 o;
#pragma unroll
  for (int e = 0; e < 8; ++e) o[e] = f2bf(acc[e]);
  *(us8*)(y + p * CDIM + c8) = o;
}

// ---------------------------------------------------------------------------
extern "C" void kernel_launch(void* const* d_in, const int* in_sizes, int n_in,
                              void* d_out, int out_size, void* d_ws, size_t ws_size,
                              hipStream_t stream) {
  (void)in_sizes; (void)n_in; (void)out_size; (void)ws_size;
  const float* x      = (const float*)d_in[0];
  const float* W_attn = (const float*)d_in[1];   // [384, 54]
  const float* b_attn = (const float*)d_in[2];
  const float* W_v    = (const float*)d_in[3];   // [384, 384]
  const float* b_v    = (const float*)d_in[4];
  const float* W_proj = (const float*)d_in[5];
  const float* b_proj = (const float*)d_in[6];

  char* ws = (char*)d_ws;
  const size_t XB   = (size_t)XELEMS * 2;                    // 38,535,168
  unsigned short* xb  = (unsigned short*)(ws);               // x bf16
  unsigned short* wvt = (unsigned short*)(ws + XB);          // Wv^T  [384,384]
  unsigned short* wpt = (unsigned short*)(ws + XB + 294912); // Wp^T  [384,384]
  unsigned short* wat = (unsigned short*)(ws + XB + 2 * 294912);            // Wa^T pad [64,384]
  float*          att = (float*)(ws + XB + 2 * 294912 + 49152);             // [50176,54] f32
  unsigned short* vb  = (unsigned short*)((char*)att + (size_t)PIX * 54 * 4);
  unsigned short* yb  = (unsigned short*)((char*)vb + XB);

  // 1) pack inputs to bf16 (+ weight transpose, attn weights zero-padded to 64)
  pack_x<<<(int)((XELEMS / 4 + 255) / 256), 256, 0, stream>>>(x, xb, XELEMS / 4);
  pack_wt<<<(CDIM * CDIM + 255) / 256, 256, 0, stream>>>(W_v,    wvt, 384, 384, 384);
  pack_wt<<<(CDIM * CDIM + 255) / 256, 256, 0, stream>>>(W_proj, wpt, 384, 384, 384);
  pack_wt<<<(64 * CDIM + 255) / 256, 256, 0, stream>>>(W_attn,  wat, 64, 54, 54);

  const dim3 blk(256);
  const dim3 gAttn(PIX / 256, 1);   // N padded to 64 -> one 64-wide column tile
  const dim3 gFull(PIX / 256, CDIM / 64);

  // 2) attention logits [50176,54] f32, then per-head softmax over 9
  gemm384_wmma<false><<<gAttn, blk, 0, stream>>>(xb, wat, b_attn, att, 54, 54);
  softmax9<<<(PIX * NHEAD + 255) / 256, 256, 0, stream>>>(att, PIX * NHEAD);

  // 3) v = x @ W_v + b_v  (bf16 out)
  gemm384_wmma<true><<<gFull, blk, 0, stream>>>(xb, wvt, b_v, vb, 384, 384);

  // 4) y = softmax-weighted 3x3 gather of v (bf16 out)
  aggregate<<<(int)((XELEMS / 8 + 255) / 256), 256, 0, stream>>>(att, vb, yb);

  // 5) out = y @ W_proj + b_proj (f32 out)
  gemm384_wmma<false><<<gFull, blk, 0, stream>>>(yb, wpt, b_proj, (float*)d_out, 384, 384);
}
